// MultiheadCrissCrossAttention_53437983097147
// MI455X (gfx1250) — matrix operand order repaired and checked by reading, not compiled, not run on hardware
//
#include <hip/hip_runtime.h>
#include <stdint.h>

// ---------------------------------------------------------------------------
// CDNA5 (gfx1250) wave32 WMMA bf16 implementation of multihead criss-cross
// attention.  All heavy math goes through v_wmma_f32_16x16x32_bf16.
// ---------------------------------------------------------------------------

typedef __bf16 bf16_t;
typedef bf16_t v16bf __attribute__((ext_vector_type(16)));
typedef float  v8f   __attribute__((ext_vector_type(8)));

#define NEG_INF (-1000000000.0f)

constexpr int Bn = 8, Cn = 512, Hn = 96, Wn = 96, NHd = 8, HDd = 64;
constexpr int Pn = Hn * Wn;                      // 9216 spatial positions
constexpr size_t NHW  = (size_t)Bn * Cn * Pn;    // 37,748,736 elements
constexpr size_t CCn  = (size_t)Cn * Cn;         // 262,144
constexpr size_t NPIX = (size_t)Bn * NHd * Hn * Wn; // 589,824

// float -> bf16 round-to-nearest-even (raw bits)
__device__ inline uint16_t f2bf(float f) {
  uint32_t u = __float_as_uint(f);
  uint32_t r = u + 0x7FFFu + ((u >> 16) & 1u);
  return (uint16_t)(r >> 16);
}

// ---- WMMA fragment loaders (ISA 7.12.2 wave32 layouts) --------------------
// A matrix 16x32 (MxK): lanes 0-15 row M=lane hold K pairs {0..7 lo / 16..23 hi},
// lanes 16-31 same rows hold K {8..15 / 24..31}.
__device__ inline v16bf frag_a_bf(const uint16_t* base, int lane, int rs) {
  int row  = lane & 15;
  int koff = (lane & 16) ? 8 : 0;
  union { uint32_t u[8]; v16bf v; } r;
  const uint16_t* p = base + row * rs;
#pragma unroll
  for (int i = 0; i < 8; ++i) {
    int k = ((i >> 2) << 4) + koff + ((i & 3) << 1);
    r.u[i] = *(const uint32_t*)(p + k);
  }
  return r.v;
}

// B matrix 32x16 (KxN) loaded from an LDS buffer stored K-innermost:
// buf[col][k] (row stride rs halves). Lanes 0-15: K {0..7 / 16..23}, lanes 16-31: K {8..15 / 24..31}.
__device__ inline v16bf frag_b_bf(const uint16_t* base, int lane, int rs) {
  int col = lane & 15;
  int kb  = (lane & 16) ? 8 : 0;
  union { uint32_t u[8]; v16bf v; } r;
  const uint16_t* p = base + col * rs + kb;
#pragma unroll
  for (int i = 0; i < 8; ++i) r.u[i] = *(const uint32_t*)(p + ((i >> 2) << 4) + ((i & 3) << 1));
  return r.v;
}

// A fragment sourced from an f32 LDS buffer (exp(S-m) scores), converted to bf16.
__device__ inline v16bf frag_a_f32(const float* base, int lane, int rs) {
  int row  = lane & 15;
  int koff = (lane & 16) ? 8 : 0;
  union { uint16_t h[16]; v16bf v; } r;
  const float* p = base + row * rs;
#pragma unroll
  for (int i = 0; i < 8; ++i) {
    int k = ((i >> 2) << 4) + koff + ((i & 3) << 1);
    r.h[2 * i]     = f2bf(p[k]);
    r.h[2 * i + 1] = f2bf(p[k + 1]);
  }
  return r.v;
}

__device__ inline v8f wmma_bf16(v16bf a, v16bf b, v8f c) {
  return __builtin_amdgcn_wmma_f32_16x16x32_bf16(false, a, false, b, (short)0, c,
                                                 false, false);
}

// ---------------------------------------------------------------------------
__global__ void cast_f32_bf16(const float* __restrict__ src,
                              uint16_t* __restrict__ dst, size_t n) {
  size_t i = (size_t)blockIdx.x * blockDim.x + threadIdx.x;
  size_t stride = (size_t)gridDim.x * blockDim.x;
  for (; i < n; i += stride) dst[i] = f2bf(src[i]);
}

// ---------------------------------------------------------------------------
// GEMM: Y[co,p] = sum_ci W[co,ci] * X[b,ci,p] + bias[co]
// MODE 0 (QKV): write bf16 into column layout [bn][w][h][d] and row layout
//               [bn][h][w][d] for the attention stage.
// MODE 1 (OUT): write f32  out = acc + bias + residual (natural layout).
// Block: 128 threads (4 waves), tile 64(M) x 64(N), K step 32.
// ---------------------------------------------------------------------------
template <int MODE>
__global__ __launch_bounds__(128) void gemm_proj(
    const uint16_t* __restrict__ Wb,    // [512][512] bf16
    const uint16_t* __restrict__ Xb,    // [B][512][9216] bf16
    const float*    __restrict__ bias,  // [512]
    uint16_t* __restrict__ colbuf,      // MODE 0
    uint16_t* __restrict__ rowbuf,      // MODE 0
    const float* __restrict__ resid,    // MODE 1
    float* __restrict__ outp)           // MODE 1
{
  __shared__ uint16_t As[64 * 32];      // [m][k]
  __shared__ uint16_t Bs[64 * 32];      // [n][k] (transposed stage)
  const int b      = blockIdx.z;
  const int tile_m = blockIdx.y * 64;
  const int tile_n = blockIdx.x * 64;
  const int tid  = threadIdx.x;
  const int lane = tid & 31, wave = tid >> 5;
  const int wm = wave >> 1, wn = wave & 1;
  const uint16_t* Bg = Xb + (size_t)b * Cn * Pn;

  v8f acc[2][2] = {};
  for (int kk = 0; kk < Cn; kk += 32) {
    // stage A tile: 64x32 halves, 256 uint4
#pragma unroll
    for (int it = 0; it < 2; ++it) {
      int idx = tid + 128 * it;
      int row = (idx * 8) >> 5, k = (idx * 8) & 31;
      *(uint4*)(As + idx * 8) =
          *(const uint4*)(Wb + (size_t)(tile_m + row) * Cn + kk + k);
    }
    // stage B tile transposed: read [k][n] coalesced, store [n][k]
#pragma unroll
    for (int it = 0; it < 2; ++it) {
      int idx = tid + 128 * it;
      int n8 = idx & 7, k = idx >> 3;
      union { uint4 q; uint16_t h[8]; } u;
      u.q = *(const uint4*)(Bg + (size_t)(kk + k) * Pn + tile_n + n8 * 8);
#pragma unroll
      for (int i = 0; i < 8; ++i) Bs[(n8 * 8 + i) * 32 + k] = u.h[i];
    }
    __syncthreads();
    v16bf a0 = frag_a_bf(As + (wm * 32 + 0)  * 32, lane, 32);
    v16bf a1 = frag_a_bf(As + (wm * 32 + 16) * 32, lane, 32);
    v16bf b0 = frag_b_bf(Bs + (wn * 32 + 0)  * 32, lane, 32);
    v16bf b1 = frag_b_bf(Bs + (wn * 32 + 16) * 32, lane, 32);
    acc[0][0] = wmma_bf16(a0, b0, acc[0][0]);
    acc[0][1] = wmma_bf16(a0, b1, acc[0][1]);
    acc[1][0] = wmma_bf16(a1, b0, acc[1][0]);
    acc[1][1] = wmma_bf16(a1, b1, acc[1][1]);
    __syncthreads();
  }

  const int col   = lane & 15;
  const int rbase = (lane & 16) ? 8 : 0;
#pragma unroll
  for (int i = 0; i < 2; ++i) {
#pragma unroll
    for (int j = 0; j < 2; ++j) {
      int co0 = tile_m + wm * 32 + i * 16;
      int p   = tile_n + wn * 32 + j * 16 + col;
      int h = p / Wn, w = p % Wn;
#pragma unroll
      for (int v = 0; v < 8; ++v) {
        int co = co0 + rbase + v;
        float val = acc[i][j][v] + bias[co];
        if (MODE == 0) {
          int n = co >> 6, d = co & 63;
          size_t bn = (size_t)b * NHd + n;
          uint16_t bf = f2bf(val);
          colbuf[((bn * Wn + w) * Hn + h) * HDd + d] = bf;
          rowbuf[((bn * Hn + h) * Wn + w) * HDd + d] = bf;
        } else {
          size_t o = ((size_t)b * Cn + co) * Pn + p;
          outp[o] = val + resid[o];
        }
      }
    }
  }
}

// ---------------------------------------------------------------------------
// One workgroup per attention line (a column for COLMODE, a row otherwise).
// S = Q K^T (96x96, K=64) -> masked max/sum -> P = exp(S-m) -> O = P V.
// Q/K/V line blocks are contiguous 96x64 bf16. 128 threads (4 waves).
// ---------------------------------------------------------------------------
template <bool COLMODE>
__global__ __launch_bounds__(128) void attn_line(
    const uint16_t* __restrict__ Qb, const uint16_t* __restrict__ Kb,
    const uint16_t* __restrict__ Vb, float* __restrict__ Obuf,
    float* __restrict__ mstat, float* __restrict__ sstat)
{
  __shared__ uint16_t Qs[Hn * HDd];   // 96x64 Q line
  __shared__ uint16_t Ks[Hn * HDd];   // 96x64 K line, later Vt (64x96)
  __shared__ float    Ss[Hn * Hn];    // scores then exp(S - m)
  const int fix = blockIdx.x;                       // w (COL) or h (ROW)
  const int bn  = blockIdx.z * NHd + blockIdx.y;
  const int tid = threadIdx.x;
  const int lane = tid & 31, wave = tid >> 5;
  const int wm = wave >> 1, wn = wave & 1;
  const size_t lineoff = ((size_t)bn * 96 + fix) * (96 * 64);
  const uint16_t* Qg = Qb + lineoff;
  const uint16_t* Kg = Kb + lineoff;
  const uint16_t* Vg = Vb + lineoff;

  // stage Q and K (768 uint4 each)
#pragma unroll
  for (int it = 0; it < 6; ++it) {
    int idx = tid + 128 * it;
    ((uint4*)Qs)[idx] = ((const uint4*)Qg)[idx];
    ((uint4*)Ks)[idx] = ((const uint4*)Kg)[idx];
  }
  __syncthreads();

  // S = Q * K^T : 2x2 wave grid, 3x3 tiles of 16x16 each, K loop 64
  v8f sacc[3][3] = {};
  for (int kk = 0; kk < 64; kk += 32) {
    v16bf af[3], bv[3];
#pragma unroll
    for (int t = 0; t < 3; ++t)
      af[t] = frag_a_bf(Qs + (wm * 48 + t * 16) * 64 + kk, lane, 64);
#pragma unroll
    for (int t = 0; t < 3; ++t)
      bv[t] = frag_b_bf(Ks + (wn * 48 + t * 16) * 64 + kk, lane, 64);
#pragma unroll
    for (int i = 0; i < 3; ++i)
#pragma unroll
      for (int j = 0; j < 3; ++j)
        sacc[i][j] = wmma_bf16(af[i], bv[j], sacc[i][j]);
  }
  {
    int col = lane & 15, rb = (lane & 16) ? 8 : 0;
#pragma unroll
    for (int i = 0; i < 3; ++i)
#pragma unroll
      for (int j = 0; j < 3; ++j) {
        int r0 = wm * 48 + i * 16, c0 = wn * 48 + j * 16 + col;
#pragma unroll
        for (int v = 0; v < 8; ++v) Ss[(r0 + rb + v) * 96 + c0] = sacc[i][j][v];
      }
  }
  __syncthreads();

  // overwrite Ks with V transposed: Vt[d][pos]
#pragma unroll
  for (int it = 0; it < 6; ++it) {
    int idx = tid + 128 * it;
    int d8 = idx & 7, pos = idx >> 3;
    union { uint4 q; uint16_t h8[8]; } u;
    u.q = ((const uint4*)Vg)[idx];
#pragma unroll
    for (int i = 0; i < 8; ++i) Ks[(d8 * 8 + i) * 96 + pos] = u.h8[i];
  }
  // per-row softmax stats + exp in place (one row per thread)
  if (tid < 96) {
    int r = tid;
    float m = -3.0e38f;
    for (int k = 0; k < 96; ++k) {
      float sv = (COLMODE && k == r) ? NEG_INF : Ss[r * 96 + k];
      m = fmaxf(m, sv);
    }
    float s = 0.f;
    for (int k = 0; k < 96; ++k) {
      float sv = (COLMODE && k == r) ? NEG_INF : Ss[r * 96 + k];
      float e = __expf(sv - m);
      s += e;
      Ss[r * 96 + k] = e;
    }
    size_t sidx = COLMODE ? ((size_t)bn * 96 + r) * 96 + fix
                          : ((size_t)bn * 96 + fix) * 96 + r;
    mstat[sidx] = m;
    sstat[sidx] = s;
  }
  __syncthreads();

  // O = P(96x96) * V(96x64): 2x2 wave grid, 3x2 tiles each, K loop 96
  v8f oacc[3][2] = {};
  for (int kk = 0; kk < 96; kk += 32) {
    v16bf af[3], bv[2];
#pragma unroll
    for (int t = 0; t < 3; ++t)
      af[t] = frag_a_f32(Ss + (wm * 48 + t * 16) * 96 + kk, lane, 96);
#pragma unroll
    for (int t = 0; t < 2; ++t)
      bv[t] = frag_b_bf(Ks + (wn * 32 + t * 16) * 96 + kk, lane, 96);
#pragma unroll
    for (int i = 0; i < 3; ++i)
#pragma unroll
      for (int j = 0; j < 2; ++j)
        oacc[i][j] = wmma_bf16(af[i], bv[j], oacc[i][j]);
  }
  float* Og = Obuf + lineoff;
  {
    int col = lane & 15, rb = (lane & 16) ? 8 : 0;
#pragma unroll
    for (int i = 0; i < 3; ++i)
#pragma unroll
      for (int j = 0; j < 2; ++j) {
        int r0 = wm * 48 + i * 16, d0 = wn * 32 + j * 16 + col;
#pragma unroll
        for (int v = 0; v < 8; ++v)
          Og[(size_t)(r0 + rb + v) * 64 + d0] = oacc[i][j][v];
      }
  }
}

// ---------------------------------------------------------------------------
// Exact merge of the two partial softmaxes (concat over 192 logits), gamma
// scale, bf16 store in natural [b][c][p] layout for the output projection.
// ---------------------------------------------------------------------------
__global__ void merge_out(const float* __restrict__ O1,   // [bn][w][h][d]
                          const float* __restrict__ O2,   // [bn][h][w][d]
                          const float* __restrict__ m1s,
                          const float* __restrict__ s1s,
                          const float* __restrict__ m2s,
                          const float* __restrict__ s2s,
                          const float* __restrict__ gamma,
                          uint16_t* __restrict__ aout)
{
  size_t pid = (size_t)blockIdx.x * blockDim.x + threadIdx.x;
  if (pid >= NPIX) return;
  int bn  = (int)(pid / Pn);
  int rem = (int)(pid % Pn);
  int h = rem / Wn, w = rem % Wn;
  float m1 = m1s[pid], s1 = s1s[pid], m2 = m2s[pid], s2 = s2s[pid];
  float m  = fmaxf(m1, m2);
  float w1 = __expf(m1 - m), w2 = __expf(m2 - m);
  float inv = gamma[0] / (s1 * w1 + s2 * w2);
  const float4* o1 = (const float4*)(O1 + (((size_t)bn * Wn + w) * Hn + h) * HDd);
  const float4* o2 = (const float4*)(O2 + (((size_t)bn * Hn + h) * Wn + w) * HDd);
  int b = bn >> 3, n = bn & 7;
  size_t base = ((size_t)b * Cn + n * HDd) * Pn + rem;
#pragma unroll
  for (int q = 0; q < 16; ++q) {
    float4 a = o1[q], c = o2[q];
    aout[base + (size_t)(q * 4 + 0) * Pn] = f2bf((a.x * w1 + c.x * w2) * inv);
    aout[base + (size_t)(q * 4 + 1) * Pn] = f2bf((a.y * w1 + c.y * w2) * inv);
    aout[base + (size_t)(q * 4 + 2) * Pn] = f2bf((a.z * w1 + c.z * w2) * inv);
    aout[base + (size_t)(q * 4 + 3) * Pn] = f2bf((a.w * w1 + c.w * w2) * inv);
  }
}

// ---------------------------------------------------------------------------
extern "C" void kernel_launch(void* const* d_in, const int* in_sizes, int n_in,
                              void* d_out, int out_size, void* d_ws,
                              size_t ws_size, hipStream_t stream) {
  const float* x     = (const float*)d_in[0];
  const float* wq    = (const float*)d_in[1];
  const float* bq    = (const float*)d_in[2];
  const float* wk    = (const float*)d_in[3];
  const float* bk    = (const float*)d_in[4];
  const float* wv    = (const float*)d_in[5];
  const float* bv    = (const float*)d_in[6];
  const float* wo    = (const float*)d_in[7];
  const float* bo    = (const float*)d_in[8];
  const float* gamma = (const float*)d_in[9];
  float* outp = (float*)d_out;

  // workspace layout (bf16 region first, then f32 region) ~0.88 GB
  uint16_t* p16 = (uint16_t*)d_ws;
  uint16_t* xb  = p16; p16 += NHW;
  uint16_t* wqb = p16; p16 += CCn;
  uint16_t* wkb = p16; p16 += CCn;
  uint16_t* wvb = p16; p16 += CCn;
  uint16_t* wob = p16; p16 += CCn;
  uint16_t* Qc  = p16; p16 += NHW;
  uint16_t* Kc  = p16; p16 += NHW;
  uint16_t* Vc  = p16; p16 += NHW;
  uint16_t* Qr  = p16; p16 += NHW;
  uint16_t* Kr  = p16; p16 += NHW;
  uint16_t* Vr  = p16; p16 += NHW;
  uint16_t* aout = p16; p16 += NHW;
  float* pf  = (float*)p16;
  float* O1  = pf; pf += NHW;
  float* O2  = pf; pf += NHW;
  float* m1s = pf; pf += NPIX;
  float* s1s = pf; pf += NPIX;
  float* m2s = pf; pf += NPIX;
  float* s2s = pf; pf += NPIX;

  cast_f32_bf16<<<4096, 256, 0, stream>>>(x,  xb,  NHW);
  cast_f32_bf16<<<512,  256, 0, stream>>>(wq, wqb, CCn);
  cast_f32_bf16<<<512,  256, 0, stream>>>(wk, wkb, CCn);
  cast_f32_bf16<<<512,  256, 0, stream>>>(wv, wvb, CCn);
  cast_f32_bf16<<<512,  256, 0, stream>>>(wo, wob, CCn);

  dim3 gg(Pn / 64, Cn / 64, Bn);  // (144, 8, 8)
  gemm_proj<0><<<gg, 128, 0, stream>>>(wqb, xb, bq, Qc, Qr, nullptr, nullptr);
  gemm_proj<0><<<gg, 128, 0, stream>>>(wkb, xb, bk, Kc, Kr, nullptr, nullptr);
  gemm_proj<0><<<gg, 128, 0, stream>>>(wvb, xb, bv, Vc, Vr, nullptr, nullptr);

  dim3 ga(96, NHd, Bn);           // one WG per attention line
  attn_line<true ><<<ga, 128, 0, stream>>>(Qc, Kc, Vc, O1, m1s, s1s);
  attn_line<false><<<ga, 128, 0, stream>>>(Qr, Kr, Vr, O2, m2s, s2s);

  merge_out<<<(unsigned)((NPIX + 255) / 256), 256, 0, stream>>>(
      O1, O2, m1s, s1s, m2s, s2s, gamma, aout);

  gemm_proj<1><<<gg, 128, 0, stream>>>(wob, aout, bo, nullptr, nullptr, x, outp);
}
